// LowLevelModule_4105988735055
// MI455X (gfx1250) — matrix operand order repaired and checked by reading, not compiled, not run on hardware
//
#include <hip/hip_runtime.h>
#include <hip/hip_bf16.h>
#include <math.h>

// ---- problem constants (match reference) ----
#define BB   1024
#define SS   64
#define II   2048
#define HH   2048
#define HHI  2048
#define LL   4
#define NHH  16
#define HDD  128   // HH / NHH

typedef __attribute__((ext_vector_type(16))) __bf16 v16bf;
typedef __attribute__((ext_vector_type(8)))  float  v8f;

__device__ __forceinline__ float dev_sigmoid(float x) { return 1.0f / (1.0f + __expf(-x)); }
__device__ __forceinline__ float dev_gelu(float x)    { return 0.5f * x * (1.0f + erff(x * 0.70710678118654752f)); }

// ============================================================================
// bf16 WMMA GEMM:  C[M,N] = A[M,K](f32->bf16) @ B[K,N](f32->bf16) + bias, epilogue
//
// Block = 256 threads (8 waves). One block computes a 64(M) x 512(N) tile:
// each wave owns a 64x64 sub-tile (4x4 grid of 16x16 WMMA tiles).
// The shared 64x32 A tile per k-step is staged in LDS via the CDNA5 async
// DMA path (global_load_async_to_lds_b128 / ASYNCcnt), double-buffered so the
// DMA of step s+1 overlaps the WMMA of step s. B (weights) streams from
// global with a prefetch hint for the next k-step.
//
// EPI: 0=f32 out (+bias)        1=f32 out (+bias + ts_emb row)
//      2=bf16 out (+bias)       3=f32 out (+bias + residual)
//      4=f32 sigmoid (+bias)    5=f32 gelu (+bias)
// ============================================================================
template<int EPI>
__global__ __launch_bounds__(256)
void gemm_wmma_bf16(const float* __restrict__ A, int lda,
                    const float* __restrict__ Bw, int ldb,
                    const float* __restrict__ bias,
                    const float* __restrict__ ts_base, const int* __restrict__ tstep,
                    const float* __restrict__ resid,
                    float* __restrict__ outF, __bf16* __restrict__ outB,
                    int M, int N, int K)
{
    __shared__ float smA[2][64 * 32];   // double-buffered 64x32 f32 A tile (2 x 8KB)

    const int wave = threadIdx.x >> 5;
    const int lane = threadIdx.x & 31;
    const int half = lane >> 4;
    const int l16  = lane & 15;

    const int tnGroups = (N >> 6) >> 3;                 // blocks per tile-row
    const int tm = blockIdx.x / tnGroups;
    const int tn = (blockIdx.x - tm * tnGroups) * 8 + wave;
    const int rowBase = tm << 6;                        // shared by all 8 waves
    const int colBase = tn << 6;

    // per-thread async-DMA chunk coordinates (512 x 16B chunks per tile)
    const int c0row = threadIdx.x >> 3;                 // chunk row for chunk id = tid
    const int c0col = (threadIdx.x & 7) * 4;            // f32 column within row
    // second chunk: id = tid + 256 -> row += 32
    auto issueA = [&](int buf, int kk) {
        #pragma unroll
        for (int i = 0; i < 2; ++i) {
            const int row = c0row + i * 32;
            const float* g = A + (size_t)(rowBase + row) * lda + kk + c0col;
            const unsigned ldsOff =
                (unsigned)(size_t)(const void*)&smA[buf][row * 32 + c0col];
            asm volatile("global_load_async_to_lds_b128 %0, %1, off"
                         :: "v"(ldsOff), "v"((unsigned long long)(size_t)g)
                         : "memory");
        }
    };

    // accumulator init with bias (all 8 f32 elems of a lane share one column)
    v8f acc[4][4];
    const float* b2 = nullptr;
    if (EPI == 1) b2 = ts_base + (size_t)tstep[0] * N;
    #pragma unroll
    for (int ni = 0; ni < 4; ++ni) {
        const int col = colBase + ni * 16 + l16;
        float bv = bias ? bias[col] : 0.0f;
        if (EPI == 1) bv += b2[col];
        #pragma unroll
        for (int mi = 0; mi < 4; ++mi)
            #pragma unroll
            for (int v = 0; v < 8; ++v) acc[mi][ni][v] = bv;
    }

    const int nsteps = K >> 5;
    issueA(0, 0);

    for (int s = 0; s < nsteps; ++s) {
        const int kk = s << 5;
        // our async chunks for the current buffer have landed
        asm volatile("s_wait_asynccnt 0x0" ::: "memory");
        // all waves have waited -> whole tile visible; all done reading other buf
        __syncthreads();
        if (s + 1 < nsteps) issueA((s + 1) & 1, kk + 32);

        const float* __restrict__ smbuf = smA[s & 1];

        // A fragments from LDS: 16x32, contiguous k = 16*half + element (same order as B)
        v16bf afrag[4];
        #pragma unroll
        for (int mi = 0; mi < 4; ++mi) {
            const float* arow = smbuf + (mi * 16 + l16) * 32 + half * 16;
            #pragma unroll
            for (int v = 0; v < 8; ++v) {
                const int k = v * 2;
                const float2 f = *reinterpret_cast<const float2*>(arow + k);
                afrag[mi][2 * v]     = (__bf16)f.x;
                afrag[mi][2 * v + 1] = (__bf16)f.y;
            }
        }
        // B fragments from global: 16-bit B 32x16, K halves split across lane halves
        v16bf bfrag[4];
        #pragma unroll
        for (int ni = 0; ni < 4; ++ni) {
            const float* bcol = Bw + (size_t)(kk + half * 16) * ldb + colBase + ni * 16 + l16;
            #pragma unroll
            for (int v = 0; v < 8; ++v) {
                bfrag[ni][2 * v]     = (__bf16)bcol[(size_t)(2 * v) * ldb];
                bfrag[ni][2 * v + 1] = (__bf16)bcol[(size_t)(2 * v + 1) * ldb];
            }
            // prefetch next k-step of this weight column stream
            if (s + 1 < nsteps)
                __builtin_prefetch(bcol + (size_t)32 * ldb, 0, 1);
        }
        #pragma unroll
        for (int mi = 0; mi < 4; ++mi)
            #pragma unroll
            for (int ni = 0; ni < 4; ++ni)
                acc[mi][ni] = __builtin_amdgcn_wmma_f32_16x16x32_bf16(
                    false, afrag[mi], false, bfrag[ni], (short)0, acc[mi][ni], false, false);
    }

    // epilogue: f32 C/D layout — lane l16 = column, VGPR v -> row v + 8*half
    #pragma unroll
    for (int mi = 0; mi < 4; ++mi) {
        #pragma unroll
        for (int ni = 0; ni < 4; ++ni) {
            const int col = colBase + ni * 16 + l16;
            #pragma unroll
            for (int v = 0; v < 8; ++v) {
                const int row = rowBase + mi * 16 + half * 8 + v;
                float x = acc[mi][ni][v];
                if (EPI == 3) x += resid[(size_t)row * N + col];
                if (EPI == 4) x = dev_sigmoid(x);
                if (EPI == 5) x = dev_gelu(x);
                if (EPI == 2) outB[(size_t)row * N + col] = (__bf16)x;
                else          outF[(size_t)row * N + col] = x;
            }
        }
    }
}

// ============================================================================
// Per-head attention: one block per (b, head); S=64 threads.
// scores -> softmax -> ctx, K/V in bf16.
// ============================================================================
__global__ __launch_bounds__(64)
void attention_kernel(const float* __restrict__ q,
                      const __bf16* __restrict__ Kb,
                      const __bf16* __restrict__ Vb,
                      float* __restrict__ ctx)
{
    __shared__ float qs[HDD];
    __shared__ float sc[SS];
    __shared__ float es[SS];
    const int b   = blockIdx.x >> 4;   // / NHH
    const int h   = blockIdx.x & 15;
    const int tid = threadIdx.x;       // s index
    const size_t qoff = (size_t)b * HH + (size_t)h * HDD;

    for (int i = tid; i < HDD; i += 64) qs[i] = q[qoff + i];
    __syncthreads();

    const __bf16* krow = Kb + ((size_t)(b * SS + tid) * HH + (size_t)h * HDD);
    float a = 0.0f;
    #pragma unroll 8
    for (int d = 0; d < HDD; ++d) a += qs[d] * (float)krow[d];
    sc[tid] = a * 0.0883883476483184f;   // 1/sqrt(128)
    __syncthreads();

    float mx = -3.4e38f;
    for (int j = 0; j < SS; ++j) mx = fmaxf(mx, sc[j]);
    es[tid] = __expf(sc[tid] - mx);
    __syncthreads();

    float sum = 0.0f;
    for (int j = 0; j < SS; ++j) sum += es[j];
    const float inv = 1.0f / sum;

    float c0 = 0.0f, c1 = 0.0f;
    for (int s = 0; s < SS; ++s) {
        const float w = es[s] * inv;
        const __bf16* vrow = Vb + ((size_t)(b * SS + s) * HH + (size_t)h * HDD);
        c0 += w * (float)vrow[tid];
        c1 += w * (float)vrow[tid + 64];
    }
    ctx[qoff + tid]      = c0;
    ctx[qoff + tid + 64] = c1;
}

// ---- small elementwise kernels ----
__global__ __launch_bounds__(256)
void build_comb(const float* __restrict__ x, const float* __restrict__ h,
                float* __restrict__ comb)
{
    const int i = blockIdx.x * 256 + threadIdx.x;       // over BB*2*HH
    const int b = i >> 12;
    const int j = i & 4095;
    comb[(size_t)i] = (j < HH) ? x[(size_t)b * HH + j] : h[(size_t)b * HH + (j - HH)];
}

__global__ __launch_bounds__(256)
void mul_rh_into_comb(const float* __restrict__ r, const float* __restrict__ h,
                      float* __restrict__ comb)
{
    const int i = blockIdx.x * 256 + threadIdx.x;       // over BB*HH
    const int b = i >> 11;
    const int j = i & 2047;
    comb[(size_t)b * (2 * HH) + HH + j] = r[(size_t)i] * h[(size_t)i];
}

__global__ __launch_bounds__(256)
void new_hidden_kernel(const float* __restrict__ u, const float* __restrict__ s,
                       const float* __restrict__ cand, const float* __restrict__ h,
                       float* __restrict__ nh)
{
    const int i = blockIdx.x * 256 + threadIdx.x;       // over BB*HH
    const float uu = u[i], ss = s[i], cc = cand[i], hh = h[i];
    nh[(size_t)i] = (1.0f - uu) * hh + uu * (ss * cc + (1.0f - ss) * hh);
}

__global__ __launch_bounds__(256)
void layernorm_kernel(const float* __restrict__ x, const float* __restrict__ g,
                      const float* __restrict__ bbias, float* __restrict__ out)
{
    __shared__ float red[256];
    const int row = blockIdx.x;
    const float* xr = x + (size_t)row * HH;
    float s = 0.0f;
    for (int j = threadIdx.x; j < HH; j += 256) s += xr[j];
    red[threadIdx.x] = s; __syncthreads();
    for (int st = 128; st > 0; st >>= 1) {
        if (threadIdx.x < st) red[threadIdx.x] += red[threadIdx.x + st];
        __syncthreads();
    }
    const float mu = red[0] * (1.0f / HH);
    __syncthreads();
    float vs = 0.0f;
    for (int j = threadIdx.x; j < HH; j += 256) { const float d = xr[j] - mu; vs += d * d; }
    red[threadIdx.x] = vs; __syncthreads();
    for (int st = 128; st > 0; st >>= 1) {
        if (threadIdx.x < st) red[threadIdx.x] += red[threadIdx.x + st];
        __syncthreads();
    }
    const float inv = rsqrtf(red[0] * (1.0f / HH) + 1e-5f);
    for (int j = threadIdx.x; j < HH; j += 256)
        out[(size_t)row * HH + j] = (xr[j] - mu) * inv * g[j] + bbias[j];
}

// ============================================================================
extern "C" void kernel_launch(void* const* d_in, const int* in_sizes, int n_in,
                              void* d_out, int out_size, void* d_ws, size_t ws_size,
                              hipStream_t stream) {
    const float* x      = (const float*)d_in[0];
    const float* hist   = (const float*)d_in[2];
    const float* hidden = (const float*)d_in[3];
    const float* W_in   = (const float*)d_in[4];
    const float* b_in   = (const float*)d_in[5];
    const float* ts_emb = (const float*)d_in[6];
    const float* Wq = (const float*)d_in[7];  const float* bq = (const float*)d_in[8];
    const float* Wk = (const float*)d_in[9];  const float* bk = (const float*)d_in[10];
    const float* Wv = (const float*)d_in[11]; const float* bv = (const float*)d_in[12];
    const float* Wo = (const float*)d_in[13]; const float* bo = (const float*)d_in[14];
    const float* Wu = (const float*)d_in[15]; const float* Wr = (const float*)d_in[16];
    const float* Ws = (const float*)d_in[17]; const float* Wc = (const float*)d_in[18];
    const float* bu = (const float*)d_in[19]; const float* br = (const float*)d_in[20];
    const float* bs = (const float*)d_in[21]; const float* bc = (const float*)d_in[22];
    const float* ln_g = (const float*)d_in[23]; const float* ln_b = (const float*)d_in[24];
    const int*   tstep = (const int*)d_in[25];

    const size_t BH = (size_t)BB * HH;
    float* out  = (float*)d_out;          // [B,H]
    float* newh = out + BH;               // [L,B,H]

    // workspace carve
    char* wp = (char*)d_ws;
    auto carve = [&](size_t bytes) -> void* {
        void* p = (void*)wp;
        wp += (bytes + 255) & ~(size_t)255;
        return p;
    };
    __bf16* Kb  = (__bf16*)carve((size_t)BB * SS * HH * sizeof(__bf16));
    __bf16* Vb  = (__bf16*)carve((size_t)BB * SS * HH * sizeof(__bf16));
    float*  x1  = (float*)carve(BH * sizeof(float));
    float*  qb  = (float*)carve(BH * sizeof(float));
    float*  ctx = (float*)carve(BH * sizeof(float));
    float*  x2  = (float*)carve(BH * sizeof(float));
    float*  comb= (float*)carve(BH * 2 * sizeof(float));
    float*  ub  = (float*)carve(BH * sizeof(float));
    float*  rb  = (float*)carve(BH * sizeof(float));
    float*  sb  = (float*)carve(BH * sizeof(float));
    float*  cb  = (float*)carve(BH * sizeof(float));

    const int g1   = (BB / 64) * (HH / 64) / 8;        // 64   (M=1024, N=2048)
    const int gKV  = ((BB * SS) / 64) * (HH / 64) / 8; // 4096 (M=65536)

    // 1) x1 = x @ W_in + b_in + ts_emb[t]
    gemm_wmma_bf16<1><<<g1, 256, 0, stream>>>(x, II, W_in, HH, b_in, ts_emb, tstep,
                                              nullptr, x1, nullptr, BB, HH, II);
    // 2) q = x1 @ Wq + bq
    gemm_wmma_bf16<0><<<g1, 256, 0, stream>>>(x1, HH, Wq, HH, bq, nullptr, nullptr,
                                              nullptr, qb, nullptr, BB, HH, HH);
    // 3) K = hist @ Wk + bk (bf16) ; 4) V = hist @ Wv + bv (bf16)
    gemm_wmma_bf16<2><<<gKV, 256, 0, stream>>>(hist, HHI, Wk, HH, bk, nullptr, nullptr,
                                               nullptr, nullptr, Kb, BB * SS, HH, HHI);
    gemm_wmma_bf16<2><<<gKV, 256, 0, stream>>>(hist, HHI, Wv, HH, bv, nullptr, nullptr,
                                               nullptr, nullptr, Vb, BB * SS, HH, HHI);
    // 5) per-head attention
    attention_kernel<<<BB * NHH, 64, 0, stream>>>(qb, Kb, Vb, ctx);
    // 6) x2 = x1 + ctx @ Wo + bo
    gemm_wmma_bf16<3><<<g1, 256, 0, stream>>>(ctx, HH, Wo, HH, bo, nullptr, nullptr,
                                              x1, x2, nullptr, BB, HH, HH);

    // 7) GRU-like layers
    const int gElem  = (int)(BH / 256);        // 8192
    const int gComb  = (int)(BH * 2 / 256);    // 16384
    const float* xcur = x2;
    for (int i = 0; i < LL; ++i) {
        const float* hi = hidden + (size_t)i * BH;
        build_comb<<<gComb, 256, 0, stream>>>(xcur, hi, comb);
        const size_t woff = (size_t)i * 2 * HH * HH;
        const size_t boff = (size_t)i * HH;
        gemm_wmma_bf16<4><<<g1, 256, 0, stream>>>(comb, 2 * HH, Wu + woff, HH, bu + boff,
                                                  nullptr, nullptr, nullptr, ub, nullptr,
                                                  BB, HH, 2 * HH);
        gemm_wmma_bf16<4><<<g1, 256, 0, stream>>>(comb, 2 * HH, Wr + woff, HH, br + boff,
                                                  nullptr, nullptr, nullptr, rb, nullptr,
                                                  BB, HH, 2 * HH);
        gemm_wmma_bf16<4><<<g1, 256, 0, stream>>>(comb, 2 * HH, Ws + woff, HH, bs + boff,
                                                  nullptr, nullptr, nullptr, sb, nullptr,
                                                  BB, HH, 2 * HH);
        mul_rh_into_comb<<<gElem, 256, 0, stream>>>(rb, hi, comb);
        gemm_wmma_bf16<5><<<g1, 256, 0, stream>>>(comb, 2 * HH, Wc + woff, HH, bc + boff,
                                                  nullptr, nullptr, nullptr, cb, nullptr,
                                                  BB, HH, 2 * HH);
        float* nh = newh + (size_t)i * BH;
        new_hidden_kernel<<<gElem, 256, 0, stream>>>(ub, sb, cb, hi, nh);
        xcur = nh;
    }
    // 8) layer norm -> out
    layernorm_kernel<<<BB, 256, 0, stream>>>(xcur, ln_g, ln_b, out);
}
